// CrossAttention_27333171872199
// MI455X (gfx1250) — matrix operand order repaired and checked by reading, not compiled, not run on hardware
//
#include <hip/hip_runtime.h>

// CrossAttention fused pipeline for gfx1250 (MI455X), wave32 + WMMA f16.
// B=4, N=4096, M=1024, QD=1024, CD=768, H=8, DH=64, INNER=512.

typedef __attribute__((ext_vector_type(16))) _Float16 v16h;
typedef __attribute__((ext_vector_type(8)))  float    v8f;

union FragA { v16h v; _Float16 h[16]; unsigned u[8]; };
union FragC { v8f  v; float f[8]; };
union H4    { _Float16 h[4]; uint2 u2; };

__device__ __forceinline__ v8f wmma16(const FragA& a, const FragA& b, v8f c) {
  // D = A(16x32 f16) * B(32x16 f16) + C(16x16 f32)
  return __builtin_amdgcn_wmma_f32_16x16x32_f16(
      /*neg_a=*/false, a.v, /*neg_b=*/false, b.v,
      /*c_mod=*/(short)0, c, /*reuse_a=*/false, /*reuse_b=*/false);
}

// A-fragment K-offset for vgpr v, lane-half g:  (ISA 16-bit A 16x32 table)
__device__ __forceinline__ int a_klo(int v, int g) {
  return ((v >> 2) << 4) + (g << 3) + ((v & 3) << 1);
}
// B-fragment K-offset for vgpr v, lane-half g:  (same K order as A table)
__device__ __forceinline__ int b_k(int v, int g) { return ((v >> 2) << 4) + (g << 3) + ((v & 3) << 1); }

// ---- LDS A-tile writers: f32 source (convert) or f16 source (copy) --------
__device__ __forceinline__ void stage4(_Float16* dst, const float* src) {
  float4 s = *(const float4*)src;
  H4 t; t.h[0] = (_Float16)s.x; t.h[1] = (_Float16)s.y;
        t.h[2] = (_Float16)s.z; t.h[3] = (_Float16)s.w;
  *(uint2*)dst = t.u2;
}
__device__ __forceinline__ void stage4(_Float16* dst, const _Float16* src) {
  *(uint2*)dst = *(const uint2*)src;
}

// ---------------------------------------------------------------------------
// Generic tiled GEMM: C[Mrows x Ncols] = A[Mrows x Kdim] * Bw[Kdim x Ncols]
// 128x128x32 per 256-thread block; 8 waves of 64x32; 1 WMMA per frag per
// k-step (BK=32 == WMMA K).  OUT_MODE: 0 = f16 row-major, 1 = f16 scattered
// to Vt[b][h][d][m] (H=8, DH=64 hardcoded), 2 = f32 + bias (final output).
// All dims are multiples of the tile sizes for this problem.
// ---------------------------------------------------------------------------
template <typename AT, int OUT_MODE>
__global__ __launch_bounds__(256) void gemm_wmma_kernel(
    const AT* __restrict__ A, const float* __restrict__ Bw,
    void* __restrict__ Cout, const float* __restrict__ bias,
    int Mrows, int Kdim, int Ncols, float scale, int rowsPerBatch) {
  // stride 40 halves (80B = 20 banks): 16 consecutive rows hit 16 distinct banks.
  __shared__ _Float16 Ah[128 * 40];
  __shared__ _Float16 Bt[128 * 40];   // B tile stored transposed: Bt[n][k]

  const int tid  = threadIdx.x;
  const int lane = tid & 31;
  const int w    = tid >> 5;      // 8 waves
  const int wm   = w & 1;         // 2 waves along M
  const int wn   = w >> 1;        // 4 waves along N
  const int mp   = lane & 15;
  const int g    = lane >> 4;
  const int m0   = blockIdx.y * 128;
  const int n0   = blockIdx.x * 128;

  FragC acc[4][2];
#pragma unroll
  for (int rb = 0; rb < 4; ++rb)
#pragma unroll
    for (int nb = 0; nb < 2; ++nb)
#pragma unroll
      for (int r = 0; r < 8; ++r) acc[rb][nb].f[r] = 0.0f;

  for (int k0 = 0; k0 < Kdim; k0 += 32) {
    __syncthreads();
    // A tile: 128x32 -> LDS f16 (vec4 loads, 4 per thread)
#pragma unroll
    for (int i = 0; i < 4; ++i) {
      int idx = tid + i * 256;
      int row = idx >> 3, c4 = (idx & 7) << 2;
      stage4(&Ah[row * 40 + c4], A + (size_t)(m0 + row) * Kdim + k0 + c4);
    }
    // B tile: 32x128 f32 -> transposed f16 LDS
#pragma unroll
    for (int i = 0; i < 4; ++i) {
      int idx = tid + i * 256;
      int kr = idx >> 5, c4 = (idx & 31) << 2;
      float4 s = *(const float4*)(Bw + (size_t)(k0 + kr) * Ncols + n0 + c4);
      Bt[(c4 + 0) * 40 + kr] = (_Float16)s.x;
      Bt[(c4 + 1) * 40 + kr] = (_Float16)s.y;
      Bt[(c4 + 2) * 40 + kr] = (_Float16)s.z;
      Bt[(c4 + 3) * 40 + kr] = (_Float16)s.w;
    }
    if (k0 + 32 < Kdim) {  // global_prefetch_b8 of next k-slice
      __builtin_prefetch(A + (size_t)(m0 + (tid >> 1)) * Kdim + (k0 + 32) + (tid & 1) * 16, 0, 1);
      __builtin_prefetch(Bw + (size_t)(k0 + 32 + (tid >> 3)) * Ncols + n0 + (tid & 7) * 16, 0, 1);
    }
    __syncthreads();

    FragA af[4], bf[2];
#pragma unroll
    for (int rb = 0; rb < 4; ++rb) {
      int row = wm * 64 + rb * 16 + mp;
#pragma unroll
      for (int v = 0; v < 8; ++v)
        af[rb].u[v] = *(const unsigned*)&Ah[row * 40 + a_klo(v, g)];
    }
#pragma unroll
    for (int nb = 0; nb < 2; ++nb) {
      int col = wn * 32 + nb * 16 + mp;
#pragma unroll
      for (int v = 0; v < 8; ++v)
        bf[nb].u[v] = *(const unsigned*)&Bt[col * 40 + b_k(v, g)];
    }
#pragma unroll
    for (int rb = 0; rb < 4; ++rb)
#pragma unroll
      for (int nb = 0; nb < 2; ++nb)
        acc[rb][nb].v = wmma16(af[rb], bf[nb], acc[rb][nb].v);
  }

  // Epilogue.  C layout: element(m,n): lane = (m>=8)*16 + n, vgpr = m%8.
#pragma unroll
  for (int rb = 0; rb < 4; ++rb)
#pragma unroll
    for (int nb = 0; nb < 2; ++nb)
#pragma unroll
      for (int r = 0; r < 8; ++r) {
        int gr = m0 + wm * 64 + rb * 16 + g * 8 + r;
        int gc = n0 + wn * 32 + nb * 16 + mp;
        float val = acc[rb][nb].f[r] * scale;
        if (OUT_MODE == 0) {
          ((_Float16*)Cout)[(size_t)gr * Ncols + gc] = (_Float16)val;
        } else if (OUT_MODE == 1) {
          int b = gr / rowsPerBatch, m = gr - b * rowsPerBatch;
          int h = gc >> 6, d = gc & 63;  // H=8, DH=64
          ((_Float16*)Cout)[(((size_t)(b * 8 + h) * 64 + d) * rowsPerBatch) + m] =
              (_Float16)val;
        } else {
          ((float*)Cout)[(size_t)gr * Ncols + gc] = val + bias[gc];
        }
      }
}

// ---------------------------------------------------------------------------
// Flash attention: one (b,h) per blockIdx.y, 256 query rows per block
// (32 per wave).  Q pre-scaled by 1/sqrt(DH) in the projection.  S = Q*K^T
// and O += P*V via WMMA; online softmax stats in f32; P converted through a
// per-wave LDS buffer to re-layout C-frag -> A-frag.
// ---------------------------------------------------------------------------
__global__ __launch_bounds__(256) void attn_kernel(
    const _Float16* __restrict__ Qh, const _Float16* __restrict__ Kh,
    const _Float16* __restrict__ Vth, _Float16* __restrict__ Oh) {
  constexpr int Nq = 4096, Mk = 1024, INNER = 512, H = 8, DH = 64;
  __shared__ _Float16 pbuf[8][32 * 34];  // stride 34 halves: 4B-aligned b32 loads

  const int tid = threadIdx.x, lane = tid & 31, w = tid >> 5;
  const int mp = lane & 15, g = lane >> 4;
  const int bh = blockIdx.y, b = bh >> 3, h = bh & 7;
  const int n0 = blockIdx.x * 256 + w * 32;

  // Q A-fragments: 2 row-blocks x 2 k-slices (DH=64), loaded once.
  FragA qf[2][2];
#pragma unroll
  for (int rb = 0; rb < 2; ++rb)
#pragma unroll
    for (int ks = 0; ks < 2; ++ks) {
      const _Float16* base =
          Qh + (size_t)(b * Nq + n0 + rb * 16 + mp) * INNER + h * DH + ks * 32;
#pragma unroll
      for (int v = 0; v < 8; ++v)
        qf[rb][ks].u[v] = *(const unsigned*)(base + a_klo(v, g));
    }

  FragC o[2][4];
  float mrow[2][8], lsum[2][8];
#pragma unroll
  for (int rb = 0; rb < 2; ++rb)
#pragma unroll
    for (int r = 0; r < 8; ++r) {
      mrow[rb][r] = -1e30f;
      lsum[rb][r] = 0.0f;
#pragma unroll
      for (int nd = 0; nd < 4; ++nd) o[rb][nd].f[r] = 0.0f;
    }

  for (int ms = 0; ms < Mk; ms += 32) {
    // ---- S = Q * K^T  (rows of Kh are exactly the Bt[n][k] layout) ----
    FragC s[2][2];
#pragma unroll
    for (int rb = 0; rb < 2; ++rb)
#pragma unroll
      for (int nb = 0; nb < 2; ++nb)
#pragma unroll
        for (int r = 0; r < 8; ++r) s[rb][nb].f[r] = 0.0f;
#pragma unroll
    for (int nb = 0; nb < 2; ++nb) {
      const _Float16* kbase =
          Kh + (size_t)(b * Mk + ms + nb * 16 + mp) * INNER + h * DH;
#pragma unroll
      for (int ks = 0; ks < 2; ++ks) {
        FragA kf;
#pragma unroll
        for (int v = 0; v < 8; ++v)
          kf.u[v] = *(const unsigned*)(kbase + ks * 32 + b_k(v, g));
#pragma unroll
        for (int rb = 0; rb < 2; ++rb)
          s[rb][nb].v = wmma16(qf[rb][ks], kf, s[rb][nb].v);
      }
    }
    // ---- online softmax (row = 16 lanes of one half-wave) ----
#pragma unroll
    for (int rb = 0; rb < 2; ++rb)
#pragma unroll
      for (int r = 0; r < 8; ++r) {
        float s0 = s[rb][0].f[r], s1 = s[rb][1].f[r];
        float vmax = fmaxf(s0, s1);
#pragma unroll
        for (int off = 1; off < 16; off <<= 1)
          vmax = fmaxf(vmax, __shfl_xor(vmax, off, 32));
        float mnew = fmaxf(mrow[rb][r], vmax);
        float alpha = __expf(mrow[rb][r] - mnew);
        mrow[rb][r] = mnew;
        float p0 = __expf(s0 - mnew), p1 = __expf(s1 - mnew);
        float psum = p0 + p1;
#pragma unroll
        for (int off = 1; off < 16; off <<= 1) psum += __shfl_xor(psum, off, 32);
        lsum[rb][r] = lsum[rb][r] * alpha + psum;
#pragma unroll
        for (int nd = 0; nd < 4; ++nd) o[rb][nd].f[r] *= alpha;
        int prow = rb * 16 + g * 8 + r;
        pbuf[w][prow * 34 + mp] = (_Float16)p0;
        pbuf[w][prow * 34 + 16 + mp] = (_Float16)p1;
      }
    // ---- O += P * V  (Vt rows d are the Bt[n][k] layout over keys) ----
    FragA vf[4];
#pragma unroll
    for (int nd = 0; nd < 4; ++nd) {
      const _Float16* vb =
          Vth + ((size_t)(b * H + h) * DH + nd * 16 + mp) * Mk + ms;
#pragma unroll
      for (int v = 0; v < 8; ++v)
        vf[nd].u[v] = *(const unsigned*)(vb + b_k(v, g));
    }
#pragma unroll
    for (int rb = 0; rb < 2; ++rb) {
      FragA pf;  // per-wave LDS region, same-wave DS ordering: no barrier
#pragma unroll
      for (int v = 0; v < 8; ++v)
        pf.u[v] = *(const unsigned*)&pbuf[w][(rb * 16 + mp) * 34 + a_klo(v, g)];
#pragma unroll
      for (int nd = 0; nd < 4; ++nd)
        o[rb][nd].v = wmma16(pf, vf[nd], o[rb][nd].v);
    }
  }
  // ---- normalize and write O (f16, head-interleaved [b][n][h*DH+d]) ----
#pragma unroll
  for (int rb = 0; rb < 2; ++rb)
#pragma unroll
    for (int r = 0; r < 8; ++r) {
      float inv = 1.0f / lsum[rb][r];
      int row = n0 + rb * 16 + g * 8 + r;
#pragma unroll
      for (int nd = 0; nd < 4; ++nd)
        Oh[(size_t)(b * Nq + row) * INNER + h * DH + nd * 16 + mp] =
            (_Float16)(o[rb][nd].f[r] * inv);
    }
}

// ---------------------------------------------------------------------------
extern "C" void kernel_launch(void* const* d_in, const int* in_sizes, int n_in,
                              void* d_out, int out_size, void* d_ws,
                              size_t ws_size, hipStream_t stream) {
  const float* x   = (const float*)d_in[0];  // [4,4096,1024]
  const float* ctx = (const float*)d_in[1];  // [4,1024,768]
  const float* Wq  = (const float*)d_in[2];  // [1024,512]
  const float* Wk  = (const float*)d_in[3];  // [768,512]
  const float* Wv  = (const float*)d_in[4];  // [768,512]
  const float* Wo  = (const float*)d_in[5];  // [512,1024]
  const float* bo  = (const float*)d_in[6];  // [1024]
  float* out = (float*)d_out;                // [4,4096,1024]

  // Workspace (f16): Q 16.8MB | K 4.2MB | Vt 4.2MB | O 16.8MB  (~42MB)
  _Float16* Qh  = (_Float16*)d_ws;
  _Float16* Kh  = Qh + (size_t)16384 * 512;
  _Float16* Vth = Kh + (size_t)4096 * 512;
  _Float16* Oh  = Vth + (size_t)4096 * 512;

  // Q = x*Wq, scaled by DH^-0.5 = 0.125 (exact in f16)
  gemm_wmma_kernel<float, 0><<<dim3(4, 128), 256, 0, stream>>>(
      x, Wq, Qh, nullptr, 16384, 1024, 512, 0.125f, 0);
  // K = ctx*Wk  (row-major f16 == Bt layout for Q*K^T)
  gemm_wmma_kernel<float, 0><<<dim3(4, 32), 256, 0, stream>>>(
      ctx, Wk, Kh, nullptr, 4096, 768, 512, 1.0f, 0);
  // V = ctx*Wv, scattered to Vt[b][h][d][m]
  gemm_wmma_kernel<float, 1><<<dim3(4, 32), 256, 0, stream>>>(
      ctx, Wv, Vth, nullptr, 4096, 768, 512, 1.0f, 1024);
  // Flash attention: 16 N-tiles x 32 (b,h) pairs
  attn_kernel<<<dim3(16, 32), 256, 0, stream>>>(Qh, Kh, Vth, Oh);
  // out = O*Wo + bo (f32)
  gemm_wmma_kernel<_Float16, 2><<<dim3(8, 128), 256, 0, stream>>>(
      Oh, Wo, out, bo, 16384, 512, 1024, 1.0f, 0);
}
